// NNSiam_74801150427318
// MI455X (gfx1250) — matrix operand order repaired and checked by reading, not compiled, not run on hardware
//
#include <hip/hip_runtime.h>

#define B_ROWS 4096
#define Q_ROWS 25600
#define DIM    2048
#define BK 32
#define NK (DIM / BK)
#define LDSK 40   // padded row stride in halves (80B, 16B-aligned)

// fallback (f32-staging) tile
#define BM 128
#define BN 128
// fast (f16 + async) tile
#define BMF 256
#define BNF 128

typedef _Float16 v16h __attribute__((ext_vector_type(16)));
typedef float    v8f  __attribute__((ext_vector_type(8)));
typedef __fp16   h2   __attribute__((ext_vector_type(2)));   // matches cvt_pkrtz return type
typedef int      v4i  __attribute__((__vector_size__(16)));  // matches async-lds builtin param

union V16 { uint4 q[2]; v16h h; };

__device__ __forceinline__ uint2 pk4(float4 v) {
  h2 a = __builtin_amdgcn_cvt_pkrtz(v.x, v.y);
  h2 b = __builtin_amdgcn_cvt_pkrtz(v.z, v.w);
  return make_uint2(__builtin_bit_cast(unsigned int, a),
                    __builtin_bit_cast(unsigned int, b));
}

// monotone float -> uint mapping so unsigned min == float min
__device__ __forceinline__ unsigned int fsort(float f) {
  unsigned int u = __float_as_uint(f);
  return (u & 0x80000000u) ? ~u : (u | 0x80000000u);
}

// ---- gfx1250 async global->LDS copy (ASYNCcnt), with safe fallback ----
__device__ __forceinline__ void async_cp16(const void* g, void* l) {
#if __has_builtin(__builtin_amdgcn_global_load_async_to_lds_b128)
  __builtin_amdgcn_global_load_async_to_lds_b128(
      (__attribute__((address_space(1))) v4i*)(void*)g,
      (__attribute__((address_space(3))) v4i*)l, 0, 0);
#else
  *(uint4*)l = *(const uint4*)g;
#endif
}
__device__ __forceinline__ void async_wait0() {
#if __has_builtin(__builtin_amdgcn_s_wait_asynccnt)
  __builtin_amdgcn_s_wait_asynccnt(0);
#elif __has_builtin(__builtin_amdgcn_global_load_async_to_lds_b128)
  asm volatile("s_wait_asynccnt 0x0" ::: "memory");
#endif
}

// ---------------- ||q_j||^2, one wave32 per row ----------------
__global__ __launch_bounds__(256)
void qnorm_kernel(const float* __restrict__ queue, float* __restrict__ qnorm) {
  const int row  = blockIdx.x * 8 + (threadIdx.x >> 5);
  const int lane = threadIdx.x & 31;
  const float4* qr = (const float4*)(queue + (size_t)row * DIM);
  float s = 0.0f;
  for (int i = lane; i < DIM / 4; i += 32) {
    float4 v = qr[i];
    s += v.x * v.x + v.y * v.y + v.z * v.z + v.w * v.w;
  }
  for (int off = 16; off > 0; off >>= 1) s += __shfl_xor(s, off, 32);
  if (lane == 0) qnorm[row] = s;
}

// ---------------- init packed argmin ----------------
__global__ __launch_bounds__(256)
void init_kernel(unsigned long long* __restrict__ packed) {
  packed[blockIdx.x * 256 + threadIdx.x] = ~0ULL;
}

// ---------------- f32 -> f16 pre-pass (16 elems / thread) ----------------
__global__ __launch_bounds__(256)
void cvt_f16_kernel(const float* __restrict__ src, unsigned short* __restrict__ dst) {
  const size_t i = (size_t)blockIdx.x * 256 + threadIdx.x;
  const float4* s = (const float4*)src + i * 4;
  float4 v0 = s[0], v1 = s[1], v2 = s[2], v3 = s[3];
  uint2 p0 = pk4(v0), p1 = pk4(v1), p2 = pk4(v2), p3 = pk4(v3);
  uint4* d = (uint4*)dst + i * 2;
  d[0] = make_uint4(p0.x, p0.y, p1.x, p1.y);
  d[1] = make_uint4(p2.x, p2.y, p3.x, p3.y);
}

// ============ FAST PATH: f16 inputs, async-to-LDS staging ============
// 256x128 block, 8 waves, 64x64 per wave -> 16 WMMA per K-step
__global__ __launch_bounds__(256)
void nn_gemm_argmin_f16(const unsigned short* __restrict__ fh,
                        const unsigned short* __restrict__ qh,
                        const float* __restrict__ qnorm,
                        unsigned long long* __restrict__ packed) {
  __shared__ __align__(16) unsigned short lsA[2][BMF * LDSK];
  __shared__ __align__(16) unsigned short lsB[2][BNF * LDSK];
  __shared__ unsigned long long ldsArg[BMF];

  const int tid     = threadIdx.x;
  const int rowBase = blockIdx.x * BMF;
  const int colBase = blockIdx.y * BNF;

  ldsArg[tid] = ~0ULL;

  // staging: A -> thread owns row tid (64B = 4x16B); B -> row tid>>1, 2x16B
  const unsigned short* gA = fh + (size_t)(rowBase + tid) * DIM;
  const unsigned short* gB = qh + (size_t)(colBase + (tid >> 1)) * DIM + (tid & 1) * 16;
  unsigned short* lA = &lsA[0][0] + tid * LDSK;
  unsigned short* lB = &lsB[0][0] + (tid >> 1) * LDSK + (tid & 1) * 16;

  auto stage = [&](int ks, int buf) {
    const unsigned short* ga = gA + ks * BK;
    unsigned short* la = lA + buf * (BMF * LDSK);
#pragma unroll
    for (int i = 0; i < 4; ++i) async_cp16(ga + i * 8, la + i * 8);
    const unsigned short* gb = gB + ks * BK;
    unsigned short* lb = lB + buf * (BNF * LDSK);
#pragma unroll
    for (int i = 0; i < 2; ++i) async_cp16(gb + i * 8, lb + i * 8);
  };

  const int lane = tid & 31;
  const int wave = tid >> 5;
  const int wm = wave >> 1;   // 0..3 : 64-row group
  const int wn = wave & 1;    // 0..1 : 64-col group
  const int lm = lane & 15;
  const int kh = lane >> 4;

  v8f c[4][4] = {};

  auto compute = [&](int buf) {
    V16 aF[4], bF[4];
#pragma unroll
    for (int mi = 0; mi < 4; ++mi) {
      // A 16x32 f16: lanes 0-15 row M; v0-3=K[kh*8..+7], v4-7=K[16+kh*8..+7]
      const unsigned short* p = &lsA[buf][(wm * 64 + mi * 16 + lm) * LDSK];
      aF[mi].q[0] = *(const uint4*)(p + kh * 8);
      aF[mi].q[1] = *(const uint4*)(p + 16 + kh * 8);
    }
#pragma unroll
    for (int ni = 0; ni < 4; ++ni) {
      // B 32x16 f16: lane n = col; v0-3=K[kh*8..+7], v4-7=K[16+kh*8..+7]
      const unsigned short* p = &lsB[buf][(wn * 64 + ni * 16 + lm) * LDSK];
      bF[ni].q[0] = *(const uint4*)(p + kh * 8);
      bF[ni].q[1] = *(const uint4*)(p + 16 + kh * 8);
    }
#pragma unroll
    for (int mi = 0; mi < 4; ++mi)
#pragma unroll
      for (int ni = 0; ni < 4; ++ni)
        c[mi][ni] = __builtin_amdgcn_wmma_f32_16x16x32_f16(
            false, aF[mi].h, false, bF[ni].h, (short)0, c[mi][ni], false, false);
  };

  stage(0, 0);
  async_wait0();
  __syncthreads();
#pragma unroll 1
  for (int ks = 0; ks < NK; ++ks) {
    const int buf = ks & 1;
    if (ks + 1 < NK) stage(ks + 1, buf ^ 1);  // async into the other buffer
    compute(buf);
    if (ks + 1 < NK) {
      async_wait0();
      __syncthreads();
    }
  }

  // epilogue: score = qnorm[j] - 2*dot; pack (sortable<<32)|col; u64-min reduce
  int   cols[4];
  float qn[4];
#pragma unroll
  for (int ni = 0; ni < 4; ++ni) {
    cols[ni] = colBase + wn * 64 + ni * 16 + lm;
    qn[ni]   = qnorm[cols[ni]];
  }
#pragma unroll
  for (int mi = 0; mi < 4; ++mi) {
    const int rB = wm * 64 + mi * 16 + kh * 8;
#pragma unroll
    for (int r = 0; r < 8; ++r) {
      unsigned long long best = ~0ULL;
#pragma unroll
      for (int ni = 0; ni < 4; ++ni) {
        float v = qn[ni] - 2.0f * c[mi][ni][r];
        unsigned long long p =
            ((unsigned long long)fsort(v) << 32) | (unsigned int)cols[ni];
        best = p < best ? p : best;
      }
      atomicMin(&ldsArg[rB + r], best);
    }
  }
  __syncthreads();
  atomicMin(&packed[rowBase + tid], ldsArg[tid]);
}

// ============ FALLBACK: f32 inputs, convert-in-kernel ============
__global__ __launch_bounds__(256)
void nn_gemm_argmin(const float* __restrict__ features,
                    const float* __restrict__ queue,
                    const float* __restrict__ qnorm,
                    unsigned long long* __restrict__ packed) {
  __shared__ __align__(16) unsigned short lsA[2][BM * LDSK];
  __shared__ __align__(16) unsigned short lsB[2][BN * LDSK];
  __shared__ unsigned long long ldsArg[BM];

  const int tid     = threadIdx.x;
  const int rowBase = blockIdx.x * BM;
  const int colBase = blockIdx.y * BN;

  for (int i = tid; i < BM; i += 256) ldsArg[i] = ~0ULL;

  const int srow  = tid >> 1;
  const int shalf = tid & 1;
  const float* gA = features + (size_t)(rowBase + srow) * DIM + shalf * 16;
  const float* gB = queue    + (size_t)(colBase + srow) * DIM + shalf * 16;

  const int lane = tid & 31;
  const int wave = tid >> 5;
  const int wm = wave >> 2;
  const int wn = wave & 3;
  const int lm = lane & 15;
  const int kh = lane >> 4;

  v8f c[4][2] = {};
  float4 ra[4], rb[4];

  auto loadG = [&](int ks) {
    const float4* pa = (const float4*)(gA + ks * BK);
    const float4* pb = (const float4*)(gB + ks * BK);
    ra[0] = pa[0]; ra[1] = pa[1]; ra[2] = pa[2]; ra[3] = pa[3];
    rb[0] = pb[0]; rb[1] = pb[1]; rb[2] = pb[2]; rb[3] = pb[3];
  };

  auto storeL = [&](int buf) {
    uint2 a0 = pk4(ra[0]), a1 = pk4(ra[1]), a2 = pk4(ra[2]), a3 = pk4(ra[3]);
    uint4* dA = (uint4*)&lsA[buf][srow * LDSK + shalf * 16];
    dA[0] = make_uint4(a0.x, a0.y, a1.x, a1.y);
    dA[1] = make_uint4(a2.x, a2.y, a3.x, a3.y);
    uint2 b0 = pk4(rb[0]), b1 = pk4(rb[1]), b2 = pk4(rb[2]), b3 = pk4(rb[3]);
    uint4* dB = (uint4*)&lsB[buf][srow * LDSK + shalf * 16];
    dB[0] = make_uint4(b0.x, b0.y, b1.x, b1.y);
    dB[1] = make_uint4(b2.x, b2.y, b3.x, b3.y);
  };

  auto compute = [&](int buf) {
    V16 aF[4], bF[2];
#pragma unroll
    for (int mi = 0; mi < 4; ++mi) {
      const unsigned short* p = &lsA[buf][(wm * 64 + mi * 16 + lm) * LDSK];
      aF[mi].q[0] = *(const uint4*)(p + kh * 8);
      aF[mi].q[1] = *(const uint4*)(p + 16 + kh * 8);
    }
#pragma unroll
    for (int ni = 0; ni < 2; ++ni) {
      const unsigned short* p = &lsB[buf][(wn * 32 + ni * 16 + lm) * LDSK];
      bF[ni].q[0] = *(const uint4*)(p + kh * 8);
      bF[ni].q[1] = *(const uint4*)(p + 16 + kh * 8);
    }
#pragma unroll
    for (int mi = 0; mi < 4; ++mi)
#pragma unroll
      for (int ni = 0; ni < 2; ++ni)
        c[mi][ni] = __builtin_amdgcn_wmma_f32_16x16x32_f16(
            false, aF[mi].h, false, bF[ni].h, (short)0, c[mi][ni], false, false);
  };

  loadG(0);
  storeL(0);
  __syncthreads();
#pragma unroll 1
  for (int ks = 0; ks < NK; ++ks) {
    const int buf = ks & 1;
    if (ks + 1 < NK) loadG(ks + 1);
    compute(buf);
    if (ks + 1 < NK) {
      storeL(buf ^ 1);
      __syncthreads();
    }
  }

  const int col0 = colBase + wn * 32 + lm;
  const int col1 = col0 + 16;
  const float qn0 = qnorm[col0];
  const float qn1 = qnorm[col1];
#pragma unroll
  for (int mi = 0; mi < 4; ++mi) {
    const int rBase = wm * 64 + mi * 16 + kh * 8;
#pragma unroll
    for (int r = 0; r < 8; ++r) {
      float v0 = qn0 - 2.0f * c[mi][0][r];
      float v1 = qn1 - 2.0f * c[mi][1][r];
      unsigned long long p0 = ((unsigned long long)fsort(v0) << 32) | (unsigned int)col0;
      unsigned long long p1 = ((unsigned long long)fsort(v1) << 32) | (unsigned int)col1;
      unsigned long long pm = p0 < p1 ? p0 : p1;
      atomicMin(&ldsArg[rBase + r], pm);
    }
  }
  __syncthreads();
  for (int i = tid; i < BM; i += 256)
    atomicMin(&packed[rowBase + i], ldsArg[i]);
}

// ---------------- gather nearest queue rows ----------------
__global__ __launch_bounds__(256)
void gather_kernel(const float* __restrict__ queue,
                   const unsigned long long* __restrict__ packed,
                   float* __restrict__ out) {
  const int b = blockIdx.x;
  const unsigned int idx = (unsigned int)(packed[b] & 0xFFFFFFFFull);
  const float4* src = (const float4*)(queue + (size_t)idx * DIM);
  float4* dst = (float4*)(out + (size_t)b * DIM);
  for (int i = threadIdx.x; i < DIM / 4; i += 256) dst[i] = src[i];
}

extern "C" void kernel_launch(void* const* d_in, const int* in_sizes, int n_in,
                              void* d_out, int out_size, void* d_ws, size_t ws_size,
                              hipStream_t stream) {
  const float* features = (const float*)d_in[0];
  const float* queue    = (const float*)d_in[1];
  float* out = (float*)d_out;

  // ws layout: qnorm (Q f32) | packed (B u64) | fh (B*DIM f16) | qh (Q*DIM f16)
  const size_t qnorm_bytes  = (size_t)Q_ROWS * sizeof(float);      // 102400
  const size_t packed_bytes = (size_t)B_ROWS * 8;                  // 32768
  float* qnorm = (float*)d_ws;
  unsigned long long* packed =
      (unsigned long long*)((char*)d_ws + qnorm_bytes);
  unsigned short* fh = (unsigned short*)((char*)d_ws + qnorm_bytes + packed_bytes);
  unsigned short* qh = fh + (size_t)B_ROWS * DIM;
  const size_t need = qnorm_bytes + packed_bytes +
                      ((size_t)B_ROWS + Q_ROWS) * DIM * sizeof(unsigned short);

  qnorm_kernel<<<Q_ROWS / 8, 256, 0, stream>>>(queue, qnorm);
  init_kernel<<<B_ROWS / 256, 256, 0, stream>>>(packed);

  if (ws_size >= need) {
    cvt_f16_kernel<<<(B_ROWS * (DIM / 16)) / 256, 256, 0, stream>>>(features, fh);
    cvt_f16_kernel<<<(Q_ROWS * (DIM / 16)) / 256, 256, 0, stream>>>(queue, qh);
    dim3 grid(B_ROWS / BMF, Q_ROWS / BNF);  // x fastest -> queue panel reuse in L2
    nn_gemm_argmin_f16<<<grid, 256, 0, stream>>>(fh, qh, qnorm, packed);
  } else {
    dim3 grid(B_ROWS / BM, Q_ROWS / BN);
    nn_gemm_argmin<<<grid, 256, 0, stream>>>(features, queue, qnorm, packed);
  }

  gather_kernel<<<B_ROWS, 256, 0, stream>>>(queue, packed, out);
}
